// MultiHeadAttention_18940805775965
// MI455X (gfx1250) — matrix operand order repaired and checked by reading, not compiled, not run on hardware
//
#include <hip/hip_runtime.h>

// ---------------------------------------------------------------------------
// MHA forward for MI455X (gfx1250, wave32): WMMA f16->f32, double-buffered
// async global->LDS staging in the attention inner loop.
// ---------------------------------------------------------------------------
typedef _Float16 v16h __attribute__((ext_vector_type(16)));
typedef _Float16 v8h  __attribute__((ext_vector_type(8)));
typedef _Float16 v4h  __attribute__((ext_vector_type(4)));
typedef float    v8f  __attribute__((ext_vector_type(8)));

#define WMMA_F16(a, b, c) \
  __builtin_amdgcn_wmma_f32_16x16x32_f16(false, (a), false, (b), (short)0, (c), false, false)

static constexpr int   DM    = 1024;   // d_model
static constexpr int   NH    = 16;     // heads
static constexpr int   HD    = 64;     // head dim
static constexpr int   BB    = 4;      // batch
static constexpr int   TT    = 2048;   // seq len
static constexpr float SCALE = 0.125f; // 1/sqrt(64)
static constexpr float LOG2E = 1.44269504088896340736f;

// --- fragment load helpers --------------------------------------------------
__device__ __forceinline__ v16h ld16c(const _Float16* p) {  // 16 contiguous
  v8h a = *(const v8h*)p;
  v8h b = *(const v8h*)(p + 8);
  v16h r;
#pragma unroll
  for (int i = 0; i < 8; ++i) { r[i] = a[i]; r[i + 8] = b[i]; }
  return r;
}
__device__ __forceinline__ v16h ld16s(const _Float16* p0, const _Float16* p1) {
  v8h a = *(const v8h*)p0;
  v8h b = *(const v8h*)(p1);
  v16h r;
#pragma unroll
  for (int i = 0; i < 8; ++i) { r[i] = a[i]; r[i + 8] = b[i]; }
  return r;
}

// --- CDNA5 async global -> LDS copy (ASYNCcnt-tracked DMA into LDS) --------
__device__ __forceinline__ void async_ld_b128(void* lds_ptr, const void* gptr) {
  uint32_t lds = (uint32_t)(uintptr_t)lds_ptr;        // addr[31:0] == LDS offset
  uint64_t ga  = (uint64_t)(uintptr_t)gptr;
  asm volatile("global_load_async_to_lds_b128 %0, %1, off"
               :: "v"(lds), "v"(ga) : "memory");
}
__device__ __forceinline__ void wait_async0() {
  asm volatile("s_wait_asynccnt 0x0" ::: "memory");
}

// --- fp32 -> f16 conversion -------------------------------------------------
__global__ void __launch_bounds__(256)
cvt_f32_f16_kernel(const float* __restrict__ src, _Float16* __restrict__ dst, int n4) {
  int i = blockIdx.x * 256 + threadIdx.x;
  if (i < n4) {
    float4 v = ((const float4*)src)[i];
    v4h r = { (_Float16)v.x, (_Float16)v.y, (_Float16)v.z, (_Float16)v.w };
    ((v4h*)dst)[i] = r;
  }
}

// --- unified GEMM: OUT = X(f16, 8192x1024) @ W(f16, 1024x1024)^T ------------
// mode 0: f16 out at [b][h][t][d] (Q,K)   mode 1: f16 out at [b][h][d][t] (V^T)
// mode 2: fp32 out row-major (final projection)
// 256 thr = 8 waves as 2(M) x 4(N); wave tile 64x64; block tile 128x256.
__global__ void __launch_bounds__(256)
gemm_kernel(const _Float16* __restrict__ X, const _Float16* __restrict__ W,
            void* __restrict__ outp, int mode) {
  const int w  = threadIdx.x >> 5;
  const int L  = threadIdx.x & 31;
  const int lo = L & 15, hi = L >> 4;
  const int rowbase = blockIdx.y * 128 + (w >> 2) * 64;
  const int colbase = blockIdx.x * 256 + (w & 3) * 64;

  v8f acc[4][4] = {};
  for (int k = 0; k < DM; k += 32) {
    v16h af[4];
#pragma unroll
    for (int mt = 0; mt < 4; ++mt) {
      const _Float16* p = X + (size_t)(rowbase + mt * 16 + lo) * DM + k + 16 * hi;
      af[mt] = ld16c(p);
    }
    v16h bf[4];
#pragma unroll
    for (int nt = 0; nt < 4; ++nt)
      bf[nt] = ld16c(W + (size_t)(colbase + nt * 16 + lo) * DM + k + 16 * hi);
#pragma unroll
    for (int mt = 0; mt < 4; ++mt)
#pragma unroll
      for (int nt = 0; nt < 4; ++nt)
        acc[mt][nt] = WMMA_F16(af[mt], bf[nt], acc[mt][nt]);
  }
#pragma unroll
  for (int mt = 0; mt < 4; ++mt)
#pragma unroll
    for (int nt = 0; nt < 4; ++nt)
#pragma unroll
      for (int r = 0; r < 8; ++r) {
        int row = rowbase + mt * 16 + r + 8 * hi;  // (b,t)
        int col = colbase + nt * 16 + lo;          // (h,d)
        float v = acc[mt][nt][r];
        if (mode == 2) {
          ((float*)outp)[(size_t)row * DM + col] = v;
        } else {
          int b = row >> 11, t = row & (TT - 1);
          int h = col >> 6,  d = col & (HD - 1);
          size_t idx = (mode == 0)
                         ? ((size_t)((b * NH + h) * TT + t)) * HD + d
                         : ((size_t)((b * NH + h) * HD + d)) * TT + t;
          ((_Float16*)outp)[idx] = (_Float16)v;
        }
      }
}

// --- flash attention --------------------------------------------------------
// Block = 4 waves = 128 queries of one (b,h). 32-key K/V tiles staged into
// double-buffered LDS by async DMA (issued one tile ahead, overlapping WMMA
// compute; one barrier per iteration). Each wave: 32 queries as two 16-wide
// S^T subtiles (S^T = K x Q^T), per-lane scalar softmax stats, P^T fragment
// built with half-wave shfl_xor. O^T = V^T x P^T accumulated in f32.
__global__ void __launch_bounds__(128)
attn_kernel(const _Float16* __restrict__ Q, const _Float16* __restrict__ K,
            const _Float16* __restrict__ Vt, _Float16* __restrict__ Out) {
  __shared__ __align__(16) _Float16 Ks[2][32][72];  // 32 keys x 64 d (padded)
  __shared__ __align__(16) _Float16 Vs[2][64][40];  // 64 d x 32 keys (padded)

  const int tid = threadIdx.x;
  const int w  = tid >> 5, L = tid & 31;
  const int lo = L & 15,  hi = L >> 4;
  const int bh   = blockIdx.x >> 4;   // (b,h) 0..63
  const int qblk = blockIdx.x & 15;   // 128-query block within head
  const int qb0  = qblk * 128;
  const int qw   = qb0 + w * 32;      // this wave's first query
  const int b = bh >> 4, h = bh & 15;

  const _Float16* Qh = Q  + (size_t)bh * (TT * HD);
  const _Float16* Kh = K  + (size_t)bh * (TT * HD);
  const _Float16* Vh = Vt + (size_t)bh * (HD * TT);

  // staging decomposition: 128 threads cover a 32x64 K tile and 64x32 V^T tile
  const int kr = tid >> 3, kc = (tid & 7) * 8;   // K: 16 rows x 8 chunks x2
  const int vd = tid >> 2, vc = (tid & 3) * 8;   // V: 32 rows x 4 chunks x2

  // Q as B operand: lane = query column, halves = d
  v16h qf[2][2];
#pragma unroll
  for (int j = 0; j < 2; ++j) {
    const _Float16* qp = Qh + (size_t)(qw + 16 * j + lo) * HD;
    qf[j][0] = ld16c(qp + 16 * hi);
    qf[j][1] = ld16c(qp + 32 + 16 * hi);
  }

  v8f   o[2][4] = {};                       // O^T accs: [q-subtile][d-tile]
  float m[2] = { -1e30f, -1e30f };
  float l[2] = { 0.0f, 0.0f };
  const int kend_blk  = qb0 + 128;
  const int kend_wave = qw + 32;

  // prologue: stage tile 0 into buffer 0
  async_ld_b128(&Ks[0][kr][kc],      Kh + (size_t)kr * HD + kc);
  async_ld_b128(&Ks[0][kr + 16][kc], Kh + (size_t)(kr + 16) * HD + kc);
  async_ld_b128(&Vs[0][vd][vc],      Vh + (size_t)vd * TT + vc);
  async_ld_b128(&Vs[0][vd + 32][vc], Vh + (size_t)(vd + 32) * TT + vc);

  for (int kb = 0; kb < kend_blk; kb += 32) {
    const int cur = (kb >> 5) & 1;
    wait_async0();    // tile `cur` landed (its DMA overlapped prior compute)
    __syncthreads();  // tile visible; all waves done reading buffer cur^1

    if (kb + 32 < kend_blk) {  // stage next tile into the retired buffer
      const int nxt = cur ^ 1;
      const int kn  = kb + 32;
      async_ld_b128(&Ks[nxt][kr][kc],      Kh + (size_t)(kn + kr) * HD + kc);
      async_ld_b128(&Ks[nxt][kr + 16][kc], Kh + (size_t)(kn + kr + 16) * HD + kc);
      async_ld_b128(&Vs[nxt][vd][vc],      Vh + (size_t)vd * TT + kn + vc);
      async_ld_b128(&Vs[nxt][vd + 32][vc], Vh + (size_t)(vd + 32) * TT + kn + vc);
    }

    if (kb >= kend_wave) continue;  // wave-uniform: past causal range

#pragma unroll
    for (int j = 0; j < 2; ++j) {
      const int qg = qw + 16 * j + lo;
      // ---- scores: two S^T key-subtiles ----------------------------------
      v8f st[2];
#pragma unroll
      for (int t = 0; t < 2; ++t) {
        const int krow = 16 * t + lo;
        v16h kf0 = ld16c(&Ks[cur][krow][16 * hi]);
        v16h kf1 = ld16c(&Ks[cur][krow][32 + 16 * hi]);
        v8f s = {};
        s = WMMA_F16(kf0, qf[j][0], s);
        s = WMMA_F16(kf1, qf[j][1], s);
        st[t] = s;
      }
      // ---- scale + causal mask + online softmax --------------------------
      float mloc = -1e30f;
#pragma unroll
      for (int t = 0; t < 2; ++t)
#pragma unroll
        for (int r = 0; r < 8; ++r) {
          float v = st[t][r] * SCALE;
          int kg = kb + 16 * t + r + 8 * hi;
          if (kg > qg) v = -1e30f;
          st[t][r] = v;
          mloc = fmaxf(mloc, v);
        }
      mloc = fmaxf(mloc, __shfl_xor(mloc, 16, 32));
      float mnew  = fmaxf(m[j], mloc);
      float alpha = __builtin_amdgcn_exp2f((m[j] - mnew) * LOG2E);
      float ssum = 0.0f;
#pragma unroll
      for (int t = 0; t < 2; ++t)
#pragma unroll
        for (int r = 0; r < 8; ++r) {
          float p = __builtin_amdgcn_exp2f((st[t][r] - mnew) * LOG2E);
          st[t][r] = p;
          ssum += p;
        }
      ssum += __shfl_xor(ssum, 16, 32);
      l[j] = l[j] * alpha + ssum;
      m[j] = mnew;
#pragma unroll
      for (int dt = 0; dt < 4; ++dt)
#pragma unroll
        for (int r = 0; r < 8; ++r) o[j][dt][r] *= alpha;

      // ---- P^T B-fragment via half-wave swaps ----------------------------
      v16h pf;
#pragma unroll
      for (int r = 0; r < 8; ++r) {
        float own0 = st[0][r], oth0 = __shfl_xor(own0, 16, 32);
        float own1 = st[1][r], oth1 = __shfl_xor(own1, 16, 32);
        pf[r]     = (_Float16)(hi ? oth1 : own0);
        pf[r + 8] = (_Float16)(hi ? own1 : oth0);
      }
      // ---- O^T += V^T x P^T ----------------------------------------------
#pragma unroll
      for (int dt = 0; dt < 4; ++dt) {
        const int dr = 16 * dt + lo;
        v16h vf = ld16c(&Vs[cur][dr][16 * hi]);
        o[j][dt] = WMMA_F16(vf, pf, o[j][dt]);
      }
    }
  }

#pragma unroll
  for (int j = 0; j < 2; ++j) {
    const float linv = 1.0f / l[j];
    _Float16* op = Out + ((size_t)(b * TT + qw + 16 * j + lo)) * DM + h * HD;
#pragma unroll
    for (int dt = 0; dt < 4; ++dt)
#pragma unroll
      for (int r = 0; r < 8; ++r) {
        int d = dt * 16 + r + 8 * hi;
        op[d] = (_Float16)(o[j][dt][r] * linv);
      }
  }
}

// ---------------------------------------------------------------------------
extern "C" void kernel_launch(void* const* d_in, const int* in_sizes, int n_in,
                              void* d_out, int out_size, void* d_ws, size_t ws_size,
                              hipStream_t stream) {
  const float* x_q  = (const float*)d_in[0];
  const float* x_kv = (const float*)d_in[1];
  const float* Wq   = (const float*)d_in[2];
  const float* Wk   = (const float*)d_in[3];
  const float* Wv   = (const float*)d_in[4];
  const float* Wo   = (const float*)d_in[5];

  // workspace layout (88 MB total; At aliases consumed xq16)
  const size_t MB = (size_t)1 << 20;
  char* ws = (char*)d_ws;
  _Float16* Wq16  = (_Float16*)(ws + 0 * MB);
  _Float16* Wk16  = (_Float16*)(ws + 2 * MB);
  _Float16* Wv16  = (_Float16*)(ws + 4 * MB);
  _Float16* Wo16  = (_Float16*)(ws + 6 * MB);
  _Float16* xq16  = (_Float16*)(ws + 8 * MB);   // [B*T][1024] f16
  _Float16* xkv16 = (_Float16*)(ws + 24 * MB);  // [B*T][1024] f16
  _Float16* Qb    = (_Float16*)(ws + 40 * MB);  // [B][H][T][64]
  _Float16* Kb    = (_Float16*)(ws + 56 * MB);  // [B][H][T][64]
  _Float16* Vtb   = (_Float16*)(ws + 72 * MB);  // [B][H][64][T]
  _Float16* At    = (_Float16*)(ws + 8 * MB);   // [B][T][1024] (alias xq16)

  const int w4 = (DM * DM) / 4;            // 262144
  const int x4 = (BB * TT * DM) / 4;       // 2097152
  cvt_f32_f16_kernel<<<w4 / 256, 256, 0, stream>>>(Wq, Wq16, w4);
  cvt_f32_f16_kernel<<<w4 / 256, 256, 0, stream>>>(Wk, Wk16, w4);
  cvt_f32_f16_kernel<<<w4 / 256, 256, 0, stream>>>(Wv, Wv16, w4);
  cvt_f32_f16_kernel<<<w4 / 256, 256, 0, stream>>>(Wo, Wo16, w4);
  cvt_f32_f16_kernel<<<x4 / 256, 256, 0, stream>>>(x_q,  xq16,  x4);
  cvt_f32_f16_kernel<<<x4 / 256, 256, 0, stream>>>(x_kv, xkv16, x4);

  dim3 g(DM / 256, (BB * TT) / 128);  // (4, 64)
  gemm_kernel<<<g, 256, 0, stream>>>(xq16,  Wq16, Qb,  0);
  gemm_kernel<<<g, 256, 0, stream>>>(xkv16, Wk16, Kb,  0);
  gemm_kernel<<<g, 256, 0, stream>>>(xkv16, Wv16, Vtb, 1);

  attn_kernel<<<BB * NH * (TT / 128), 128, 0, stream>>>(Qb, Kb, Vtb, At);

  gemm_kernel<<<g, 256, 0, stream>>>(At, Wo16, d_out, 2);
}
